// BottleNeck_attention_11811160064067
// MI455X (gfx1250) — matrix operand order repaired and checked by reading, not compiled, not run on hardware
//
#include <hip/hip_runtime.h>
#include <hip/hip_bf16.h>
#include <stdint.h>

typedef __bf16 v16bf __attribute__((ext_vector_type(16)));
typedef float  v8f   __attribute__((ext_vector_type(8)));

#define BM 128
#define BN 128
#define BK 32
#define NTHREADS 256
#define A_TILE_BYTES (BM * BK * 2)
#define B_TILE_BYTES (BN * BK * 2)

// Scalar f32 -> bf16 (native cast; backend uses v_cvt_pk_bf16_f32).
__device__ __forceinline__ unsigned short f2bf(float f) {
  union { __bf16 h; unsigned short s; } c;
  c.h = (__bf16)f;
  return c.s;
}
// Packed pair in ONE v_cvt_pk_bf16_f32 (src0 -> [15:0], src1 -> [31:16]).
__device__ __forceinline__ unsigned int f2bf2(float lo, float hi) {
  unsigned int r;
  asm("v_cvt_pk_bf16_f32 %0, %1, %2" : "=v"(r) : "v"(lo), "v"(hi));
  return r;
}

union FragAB { v16bf v; uint4 u[2]; };

// ---------------------------------------------------------------------------
// f32 -> bf16 bulk convert (each thread converts 4 elements)
// ---------------------------------------------------------------------------
__global__ __launch_bounds__(256) void f32_to_bf16_kernel(
    const float* __restrict__ in, unsigned short* __restrict__ out) {
  int i = (blockIdx.x * blockDim.x + threadIdx.x) * 4;
  float4 f = *(const float4*)(in + i);
  uint2 p;
  p.x = f2bf2(f.x, f.y);
  p.y = f2bf2(f.z, f.w);
  *(uint2*)(out + i) = p;
}

// ---------------------------------------------------------------------------
// Tiled WMMA GEMM:  C[M,N] = alpha * A[M,K] * B^T + bias
//   AF32  : A fp32 in global, converted to bf16 while staging (reg-prefetch
//           double buffer); batch-gather row map (rows_per_batch/a_batch_stride).
//           B (bf16 [N][K]) also reg-prefetched.
//   !AF32 : A bf16, staged with GLOBAL_LOAD_ASYNC_TO_LDS_B128 (ASYNCcnt),
//           ping-pong LDS buffers overlapping DMA with WMMA.
//   !TRB  : B bf16 [N][K]; async-staged when !AF32.
//   TRB   : B bf16 [K][N] (vp); reg-prefetch + transpose-scatter into LDS.
//   CF32  : C written f32, else bf16.
// Block tile 128x128x32; 8 wave32s in 2(M) x 4(N); each wave 4x2 16x16 tiles.
// ---------------------------------------------------------------------------
template<bool AF32, bool TRB, bool BIAS, bool CF32>
__global__ __launch_bounds__(NTHREADS)
void gemm_wmma(const void* __restrict__ Av, const unsigned short* __restrict__ Bmat,
               const float* __restrict__ bias, void* __restrict__ Cv,
               int K, int lda, int ldb, int ldc,
               int rows_per_batch, long long a_batch_stride,
               long long a_z, long long b_z, long long c_z, float alpha)
{
  __shared__ unsigned short lds_a[2][BM * BK];
  __shared__ unsigned short lds_b[2][BN * BK];

  const int t    = threadIdx.x;
  const int lane = t & 31;
  const int wid  = t >> 5;
  const int wm   = wid >> 2;    // 0..1  (M direction, 64 rows)
  const int wn   = wid & 3;     // 0..3  (N direction, 32 cols)
  const int l    = lane & 15;
  const int h    = lane >> 4;

  const int m_base = blockIdx.y * BM;
  const int n_base = blockIdx.x * BN;
  const int z      = blockIdx.z;

  const float*          A_f = (const float*)Av + (size_t)z * (size_t)a_z;
  const unsigned short* A_b = (const unsigned short*)Av + (size_t)z * (size_t)a_z;
  const unsigned short* B   = Bmat + (size_t)z * (size_t)b_z;

  // per-thread staging coordinates (pure functions of t)
  const int a4_row = t >> 3, a4_c4 = t & 7;     // AF32 A: float4 chunks (+32/i)
  const int cc_row = t >> 2, cc_c8 = t & 3;     // bf16 b128 chunks (+64/i)
  const int tr_kr  = t >> 4, tr_c8 = t & 15;    // TRB source chunks (+16/i)

  // LDS byte offsets for async DMA (low 32 bits of generic ptr == LDS offset)
  const unsigned abase = (unsigned)(unsigned long long)(uintptr_t)&lds_a[0][0];
  const unsigned bbase = (unsigned)(unsigned long long)(uintptr_t)&lds_b[0][0];

  float4 areg[4];   // AF32 A prefetch
  uint4  breg[2];   // B prefetch (AF32 direct or TRB source)

  const int KT = K / BK;

  // ---- producers -----------------------------------------------------------
  auto produce_reg = [&](int kt) {
    const int k0 = kt * BK;
    if constexpr (AF32) {
#pragma unroll
      for (int i = 0; i < 4; ++i) {
        int row = a4_row + 32 * i;
        int grow = m_base + row;
        int bidx = grow / rows_per_batch;
        int rrow = grow - bidx * rows_per_batch;
        areg[i] = *(const float4*)(A_f + (size_t)bidx * (size_t)a_batch_stride
                                       + (size_t)rrow * (size_t)lda + k0 + a4_c4 * 4);
      }
#pragma unroll
      for (int i = 0; i < 2; ++i) {
        int row = cc_row + 64 * i;
        breg[i] = *(const uint4*)(B + (size_t)(n_base + row) * (size_t)ldb + k0 + cc_c8 * 8);
      }
    }
    if constexpr (TRB) {
#pragma unroll
      for (int i = 0; i < 2; ++i) {
        int kr = tr_kr + 16 * i;
        breg[i] = *(const uint4*)(B + (size_t)(k0 + kr) * (size_t)ldb + n_base + tr_c8 * 8);
      }
    }
  };

  auto store_regs = [&](int buf) {
    if constexpr (AF32) {
#pragma unroll
      for (int i = 0; i < 4; ++i) {
        int row = a4_row + 32 * i;
        uint2 p;
        p.x = f2bf2(areg[i].x, areg[i].y);
        p.y = f2bf2(areg[i].z, areg[i].w);
        *(uint2*)&lds_a[buf][row * BK + a4_c4 * 4] = p;
      }
#pragma unroll
      for (int i = 0; i < 2; ++i) {
        int row = cc_row + 64 * i;
        *(uint4*)&lds_b[buf][row * BK + cc_c8 * 8] = breg[i];
      }
    }
    if constexpr (TRB) {
#pragma unroll
      for (int i = 0; i < 2; ++i) {
        int kr = tr_kr + 16 * i;
        union { uint4 q; unsigned short s[8]; } uv; uv.q = breg[i];
#pragma unroll
        for (int j = 0; j < 8; ++j)
          lds_b[buf][(tr_c8 * 8 + j) * BK + kr] = uv.s[j];
      }
    }
  };

  auto issue_async = [&](int kt, int buf) {
    if constexpr (!AF32) {
      const int k0 = kt * BK;
#pragma unroll
      for (int i = 0; i < 2; ++i) {
        int row = cc_row + 64 * i;
        unsigned long long g = (unsigned long long)(uintptr_t)(
            A_b + (size_t)(m_base + row) * (size_t)lda + k0 + cc_c8 * 8);
        unsigned loff = abase + (unsigned)buf * A_TILE_BYTES
                      + (unsigned)(row * BK + cc_c8 * 8) * 2u;
        asm volatile("global_load_async_to_lds_b128 %0, %1, off"
                     :: "v"(loff), "v"(g) : "memory");
      }
      if constexpr (!TRB) {
#pragma unroll
        for (int i = 0; i < 2; ++i) {
          int row = cc_row + 64 * i;
          unsigned long long g = (unsigned long long)(uintptr_t)(
              B + (size_t)(n_base + row) * (size_t)ldb + k0 + cc_c8 * 8);
          unsigned loff = bbase + (unsigned)buf * B_TILE_BYTES
                        + (unsigned)(row * BK + cc_c8 * 8) * 2u;
          asm volatile("global_load_async_to_lds_b128 %0, %1, off"
                       :: "v"(loff), "v"(g) : "memory");
        }
      }
    }
  };

  // ---- accumulators --------------------------------------------------------
  v8f acc[4][2];
#pragma unroll
  for (int mi = 0; mi < 4; ++mi)
#pragma unroll
    for (int ni = 0; ni < 2; ++ni)
#pragma unroll
      for (int e = 0; e < 8; ++e) acc[mi][ni][e] = 0.0f;

  // ---- pipelined main loop -------------------------------------------------
  produce_reg(0);
  issue_async(0, 0);

  for (int kt = 0; kt < KT; ++kt) {
    const int cur = kt & 1;

    store_regs(cur);                       // reg-staged operands -> LDS[cur]
    if (kt + 1 < KT) produce_reg(kt + 1);  // global->reg prefetch for next tile

    if constexpr (!AF32)
      asm volatile("s_wait_asynccnt 0x0" ::: "memory");  // own async DMAs done
    __syncthreads();                        // cross-wave: LDS[cur] fully staged

    if (kt + 1 < KT) issue_async(kt + 1, 1 - cur);  // DMA next tile during WMMA

    // fragments per ISA VGPR layout
    FragAB afr[4], bfr[2];
#pragma unroll
    for (int mi = 0; mi < 4; ++mi) {
      int rs = wm * 64 + mi * 16 + l;
      afr[mi].u[0] = *(const uint4*)&lds_a[cur][rs * BK + h * 8];
      afr[mi].u[1] = *(const uint4*)&lds_a[cur][rs * BK + 16 + h * 8];
    }
#pragma unroll
    for (int ni = 0; ni < 2; ++ni) {
      int ns = wn * 32 + ni * 16 + l;
      bfr[ni].u[0] = *(const uint4*)&lds_b[cur][ns * BK + h * 8];
      bfr[ni].u[1] = *(const uint4*)&lds_b[cur][ns * BK + 16 + h * 8];
    }

#pragma unroll
    for (int mi = 0; mi < 4; ++mi)
#pragma unroll
      for (int ni = 0; ni < 2; ++ni)
        acc[mi][ni] = __builtin_amdgcn_wmma_f32_16x16x32_bf16(
            false, afr[mi].v, false, bfr[ni].v, (short)0, acc[mi][ni], false, false);

    __syncthreads();                        // LDS[cur] free for reuse at kt+2
  }

  // ---- epilogue ------------------------------------------------------------
  float*          Cf = (float*)Cv + (size_t)z * (size_t)c_z;
  unsigned short* Cb = (unsigned short*)Cv + (size_t)z * (size_t)c_z;
#pragma unroll
  for (int mi = 0; mi < 4; ++mi) {
#pragma unroll
    for (int ni = 0; ni < 2; ++ni) {
      int n = n_base + wn * 32 + ni * 16 + l;
      float bv = BIAS ? bias[n] : 0.0f;
#pragma unroll
      for (int r = 0; r < 8; ++r) {
        int m = m_base + wm * 64 + mi * 16 + h * 8 + r;
        float val = acc[mi][ni][r] * alpha + bv;
        if (CF32) Cf[(size_t)m * (size_t)ldc + n] = val;
        else      Cb[(size_t)m * (size_t)ldc + n] = f2bf(val);
      }
    }
  }
}

// ---------------------------------------------------------------------------
// Softmax over the BATCH axis (reference uses softmax(dim=0) on [8,256,4096])
// ---------------------------------------------------------------------------
__global__ __launch_bounds__(256) void softmax_batch_kernel(
    const float* __restrict__ score, unsigned short* __restrict__ attn) {
  const int QK = 256 * 4096;
  int idx = blockIdx.x * blockDim.x + threadIdx.x;  // (q,k) pair
  float s[8];
  float mx = -3.4e38f;
#pragma unroll
  for (int b = 0; b < 8; ++b) {
    s[b] = score[(size_t)b * QK + idx];
    mx = fmaxf(mx, s[b]);
  }
  float sum = 0.0f;
#pragma unroll
  for (int b = 0; b < 8; ++b) { s[b] = __expf(s[b] - mx); sum += s[b]; }
  float inv = 1.0f / sum;
#pragma unroll
  for (int b = 0; b < 8; ++b)
    attn[(size_t)b * QK + idx] = f2bf(s[b] * inv);
}

// ---------------------------------------------------------------------------
extern "C" void kernel_launch(void* const* d_in, const int* in_sizes, int n_in,
                              void* d_out, int out_size, void* d_ws, size_t ws_size,
                              hipStream_t stream) {
  const float* q  = (const float*)d_in[0];
  const float* k  = (const float*)d_in[1];
  const float* v  = (const float*)d_in[2];
  const float* Wq = (const float*)d_in[3];
  const float* bq = (const float*)d_in[4];
  const float* Wk = (const float*)d_in[5];
  const float* bk = (const float*)d_in[6];
  const float* Wv = (const float*)d_in[7];
  const float* bv = (const float*)d_in[8];
  float* out = (float*)d_out;

  const int B = 8, L = 4096, D = 1024, Q = 256;
  const size_t MB = 1024ull * 1024ull;
  char* ws = (char*)d_ws;
  unsigned short* wq_bf  = (unsigned short*)(ws + 0 * MB);    // 2 MB
  unsigned short* wk_bf  = (unsigned short*)(ws + 2 * MB);    // 2 MB
  unsigned short* wv_bf  = (unsigned short*)(ws + 4 * MB);    // 2 MB
  unsigned short* qp_bf  = (unsigned short*)(ws + 6 * MB);    // 4 MB  [B*Q, D]
  unsigned short* kp_bf  = (unsigned short*)(ws + 10 * MB);   // 64 MB [B*L, D]
  unsigned short* vp_bf  = (unsigned short*)(ws + 74 * MB);   // 64 MB [B*L, D]
  float*          score  = (float*)(ws + 138 * MB);           // 32 MB [B,Q,L]
  unsigned short* attn_bf = (unsigned short*)(ws + 170 * MB); // 16 MB [B,Q,L]

  // 1) weights f32 -> bf16
  f32_to_bf16_kernel<<<1024, 256, 0, stream>>>(Wq, wq_bf);
  f32_to_bf16_kernel<<<1024, 256, 0, stream>>>(Wk, wk_bf);
  f32_to_bf16_kernel<<<1024, 256, 0, stream>>>(Wv, wv_bf);

  // 2) projections: x @ W^T + b  (A fp32 staged->bf16, out bf16)
  gemm_wmma<true, false, true, false><<<dim3(D / BN, (B * L) / BM, 1), NTHREADS, 0, stream>>>(
      (const void*)k, wk_bf, bk, (void*)kp_bf,
      D, D, D, D, /*rows_per_batch*/B * L, 0ll, 0ll, 0ll, 0ll, 1.0f);
  gemm_wmma<true, false, true, false><<<dim3(D / BN, (B * L) / BM, 1), NTHREADS, 0, stream>>>(
      (const void*)v, wv_bf, bv, (void*)vp_bf,
      D, D, D, D, B * L, 0ll, 0ll, 0ll, 0ll, 1.0f);
  // qp: only first 256 rows of each batch (batch-gather row map)
  gemm_wmma<true, false, true, false><<<dim3(D / BN, (B * Q) / BM, 1), NTHREADS, 0, stream>>>(
      (const void*)q, wq_bf, bq, (void*)qp_bf,
      D, D, D, D, /*rows_per_batch*/Q, (long long)L * D, 0ll, 0ll, 0ll, 1.0f);

  // 3) score = (qp @ kp^T) / 32  per batch -> f32 [B,Q,L]   (async-DMA staged)
  gemm_wmma<false, false, false, true><<<dim3(L / BN, Q / BM, B), NTHREADS, 0, stream>>>(
      (const void*)qp_bf, kp_bf, nullptr, (void*)score,
      D, D, D, L, 1, 0ll,
      (long long)Q * D, (long long)L * D, (long long)Q * L, 0.03125f);

  // 4) softmax across batch axis
  softmax_batch_kernel<<<(Q * L) / 256, 256, 0, stream>>>(score, attn_bf);

  // 5) out = attn @ vp  per batch -> f32 [B,Q,D]  (A async, B transpose-stage)
  gemm_wmma<false, true, false, true><<<dim3(D / BN, Q / BM, B), NTHREADS, 0, stream>>>(
      (const void*)attn_bf, vp_bf, nullptr, (void*)out,
      L, L, D, D, 1, 0ll,
      (long long)Q * L, (long long)L * D, (long long)Q * D, 1.0f);
}
